// MAB_86148454023859
// MI455X (gfx1250) — matrix operand order repaired and checked by reading, not compiled, not run on hardware
//
#include <hip/hip_runtime.h>

// MAB (Set-Transformer multihead attention block) for MI455X / gfx1250.
// Matmuls via v_wmma_f32_16x16x32_f16; attention K/V staged through LDS with
// double-buffered global_load_async_to_lds_b128 (ASYNCcnt-tracked DMA).

typedef __attribute__((ext_vector_type(16))) _Float16 v16h;
typedef __attribute__((ext_vector_type(8)))  float    v8f;

#define NB    4
#define NSEQ  2048      // NQ == NK
#define DM    256       // DQ == DK == DV
#define NH    4
#define HD    64
#define SCALEF 0.0625f  // 1/sqrt(256)

static __device__ __forceinline__ int laneid() { return (int)(threadIdx.x & 31); }

static __device__ __forceinline__ v8f wmma32(v16h a, v16h b, v8f c) {
  // (neg_a, A, neg_b, B, c_mod, C, reuse_a, reuse_b)
  return __builtin_amdgcn_wmma_f32_16x16x32_f16(false, a, false, b, (short)0, c,
                                                false, false);
}

// ---- Fragment loaders (ISA 7.12.2 layouts) ----------------------------------
// A (16x32 f16): lane<16 -> K[kb..kb+7],K[kb+16..kb+23] with kb=0; lane>=16 kb=8
static __device__ __forceinline__ v16h fragA_f16(const _Float16* row, int dofs) {
  const int kb = (laneid() < 16) ? 0 : 8;
  const _Float16* p = row + dofs + kb;
  v16h r;
#pragma unroll
  for (int e = 0; e < 8; ++e) { r[e] = p[e]; r[8 + e] = p[16 + e]; }
  return r;
}
static __device__ __forceinline__ v16h fragA_f32(const float* row, int dofs) {
  const int kb = (laneid() < 16) ? 0 : 8;
  const float* p = row + dofs + kb;
  v16h r;
#pragma unroll
  for (int e = 0; e < 8; ++e) { r[e] = (_Float16)p[e]; r[8 + e] = (_Float16)p[16 + e]; }
  return r;
}
// B (32x16 f16): per-lane column, lane<16 -> K[0..7],K[16..23]; lane>=16 -> K[8..15],K[24..31]
static __device__ __forceinline__ v16h fragB_f16(const _Float16* row, int dofs) {
  const int kb = (laneid() < 16) ? 0 : 8;
  const _Float16* p = row + dofs + kb;
  v16h r;
#pragma unroll
  for (int e = 0; e < 8; ++e) { r[e] = p[e]; r[8 + e] = p[16 + e]; }
  return r;
}
static __device__ __forceinline__ v16h fragB_f32(const float* row, int dofs) {
  const int kb = (laneid() < 16) ? 0 : 8;
  const float* p = row + dofs + kb;
  v16h r;
#pragma unroll
  for (int e = 0; e < 8; ++e) { r[e] = (_Float16)p[e]; r[8 + e] = (_Float16)p[16 + e]; }
  return r;
}

// Reductions across the 16-lane half that shares a C/D row set
static __device__ __forceinline__ float halfmax16(float v) {
  v = fmaxf(v, __shfl_xor(v, 1, 32));
  v = fmaxf(v, __shfl_xor(v, 2, 32));
  v = fmaxf(v, __shfl_xor(v, 4, 32));
  v = fmaxf(v, __shfl_xor(v, 8, 32));
  return v;
}
static __device__ __forceinline__ float halfsum16(float v) {
  v += __shfl_xor(v, 1, 32);
  v += __shfl_xor(v, 2, 32);
  v += __shfl_xor(v, 4, 32);
  v += __shfl_xor(v, 8, 32);
  return v;
}

// Async DMA of one 16B segment from global into LDS (per-lane addresses).
// Tracked with ASYNCcnt (one count per instruction per wave).
static __device__ __forceinline__ void async_cp16(unsigned lds_off,
                                                  const _Float16* gptr) {
  asm volatile("global_load_async_to_lds_b128 %0, %1, off"
               :
               : "v"(lds_off), "v"((unsigned long long)(uintptr_t)gptr)
               : "memory");
}
static __device__ __forceinline__ unsigned lds_offset(const void* p) {
  // generic pointers to LDS carry the LDS byte offset in their low 32 bits
  return (unsigned)(uintptr_t)p;
}

// ---- Kernel 1: projections q = Q Wq^T, k = K Wk^T, v = K Wv^T ---------------
// grid (128, 4, 3), block 128.  wave -> 16 rows x 64 cols tile, K-loop 256.
__global__ __launch_bounds__(128) void proj_kernel(
    const float* __restrict__ Q, const float* __restrict__ K,
    const float* __restrict__ Wq, const float* __restrict__ Wk,
    const float* __restrict__ Wv,
    _Float16* __restrict__ qf16, float* __restrict__ qf32,
    _Float16* __restrict__ kf16, _Float16* __restrict__ vt) {
  const int wave = threadIdx.x >> 5;
  const int lane = laneid();
  const int proj = blockIdx.z;                  // 0:q 1:k 2:v
  const int m0 = (blockIdx.x * 4 + wave) * 16;  // row tile (over B*NSEQ)
  const int n0 = blockIdx.y * 64;               // col tile
  const float* X = (proj == 0) ? Q : K;
  const float* W = (proj == 0) ? Wq : ((proj == 1) ? Wk : Wv);

  const float* xrow = X + (size_t)(m0 + (lane & 15)) * DM;
  v8f acc[4] = {};
#pragma unroll
  for (int k0 = 0; k0 < DM; k0 += 32) {
    v16h a = fragA_f32(xrow, k0);
#pragma unroll
    for (int ct = 0; ct < 4; ++ct) {
      v16h b = fragB_f32(W + (size_t)(n0 + ct * 16 + (lane & 15)) * DM, k0);
      acc[ct] = wmma32(a, b, acc[ct]);
    }
  }
  const int rbase = (lane < 16) ? 0 : 8;
  const int col = lane & 15;
#pragma unroll
  for (int ct = 0; ct < 4; ++ct) {
#pragma unroll
    for (int r = 0; r < 8; ++r) {
      const int m = m0 + rbase + r;            // global row in [0, B*NSEQ)
      const int n = n0 + ct * 16 + col;        // feature in [0, 256)
      const float val = acc[ct][r];
      const size_t idx = (size_t)m * DM + n;
      if (proj == 0) {
        qf16[idx] = (_Float16)val;
        qf32[idx] = val;
      } else if (proj == 1) {
        kf16[idx] = (_Float16)val;
      } else {
        const int b = m >> 11, j = m & 2047;   // NSEQ = 2048
        vt[((size_t)b * DM + n) * NSEQ + j] = (_Float16)val;  // v transposed
      }
    }
  }
}

// ---- Kernel 2: flash attention + residual q ---------------------------------
// grid (NSEQ/64, NH, NB), block 128 (4 waves; all share (b,h) so K/V chunks are
// staged into LDS once per block via double-buffered async DMA).
__global__ __launch_bounds__(128) void attn_kernel(
    const _Float16* __restrict__ qf16, const _Float16* __restrict__ kf16,
    const _Float16* __restrict__ vt, const float* __restrict__ qf32,
    float* __restrict__ Oattn, _Float16* __restrict__ Of16) {
  // K chunk: 32 keys x 64 d (rows padded to 72 halves = 144B, 16B aligned)
  // V chunk: 64 d x 32 keys (rows padded to 40 halves = 80B, 16B aligned)
  __shared__ __align__(16) _Float16 kbuf[2][32][72];
  __shared__ __align__(16) _Float16 vbuf[2][64][40];
  __shared__ __align__(16) _Float16 plds[4][16][40];  // P staging per wave

  const int wave = threadIdx.x >> 5;
  const int lane = laneid();
  const int tid = threadIdx.x;
  const int b = blockIdx.z, h = blockIdx.y;
  const int m0 = (blockIdx.x * 4 + wave) * 16;
  const int kb8 = (lane < 16) ? 0 : 8;
  const int rbase = (lane < 16) ? 0 : 8;
  const int col = lane & 15;

  const _Float16* kbase = kf16 + (size_t)b * NSEQ * DM + h * HD;
  const _Float16* vbase = vt + ((size_t)b * DM + h * HD) * NSEQ;

  // q A-fragments (K = 0..31, 32..63 of this head), loaded once from global
  const _Float16* qrow =
      qf16 + (size_t)(b * NSEQ + m0 + (lane & 15)) * DM + h * HD;
  const v16h qa0 = fragA_f16(qrow, 0);
  const v16h qa1 = fragA_f16(qrow, 32);

  v8f acc[4] = {};
  float mrow[8], lrow[8];
#pragma unroll
  for (int r = 0; r < 8; ++r) { mrow[r] = -__builtin_inff(); lrow[r] = 0.f; }

  // Issue async copies for chunk at key offset c0 into buffer nb.
  // 512 x 16B segments / 128 threads = 4 async b128 per thread (4 per wave).
  auto issue_chunk = [&](int c0, int nb) {
#pragma unroll
    for (int i = 0; i < 2; ++i) {          // K: 256 segs, 8 per 64-half row
      const int u = tid * 2 + i;
      const int row = u >> 3, seg = u & 7;
      async_cp16(lds_offset(&kbuf[nb][row][seg * 8]),
                 kbase + (size_t)(c0 + row) * DM + seg * 8);
    }
#pragma unroll
    for (int i = 0; i < 2; ++i) {          // V: 256 segs, 4 per 32-half row
      const int u = tid * 2 + i;
      const int row = u >> 2, seg = u & 3;
      async_cp16(lds_offset(&vbuf[nb][row][seg * 8]),
                 vbase + (size_t)row * NSEQ + c0 + seg * 8);
    }
  };

  constexpr int NCHUNK = NSEQ / 32;  // 64
  issue_chunk(0, 0);

  for (int ci = 0; ci < NCHUNK; ++ci) {
    const int cur = ci & 1;
    if (ci + 1 < NCHUNK) {
      issue_chunk((ci + 1) * 32, cur ^ 1);  // prefetch next chunk's DMA
      asm volatile("s_wait_asynccnt 0x4" ::: "memory");  // drain current 4
    } else {
      asm volatile("s_wait_asynccnt 0x0" ::: "memory");
    }
    __syncthreads();  // all waves' copies for `cur` complete

    // S = q k^T for 32 keys: two 16-key tiles, K=64 split into 2x32
    const _Float16* kr0 = &kbuf[cur][col][0];
    const _Float16* kr1 = &kbuf[cur][16 + col][0];
    v8f s0 = {}, s1 = {};
    s0 = wmma32(qa0, fragB_f16(kr0, 0), s0);
    s0 = wmma32(qa1, fragB_f16(kr0, 32), s0);
    s1 = wmma32(qa0, fragB_f16(kr1, 0), s1);
    s1 = wmma32(qa1, fragB_f16(kr1, 32), s1);

    // online softmax over the 32-key chunk (row stats live in the 16-lane half)
#pragma unroll
    for (int r = 0; r < 8; ++r) {
      const float a0 = s0[r] * SCALEF;
      const float a1 = s1[r] * SCALEF;
      const float cm = halfmax16(fmaxf(a0, a1));
      const float mn = fmaxf(mrow[r], cm);
      const float alpha = __expf(mrow[r] - mn);
      const float p0 = __expf(a0 - mn);
      const float p1 = __expf(a1 - mn);
      lrow[r] = lrow[r] * alpha + halfsum16(p0 + p1);
      mrow[r] = mn;
#pragma unroll
      for (int dt = 0; dt < 4; ++dt) acc[dt][r] *= alpha;
      plds[wave][rbase + r][col] = (_Float16)p0;       // key c0+col
      plds[wave][rbase + r][16 + col] = (_Float16)p1;  // key c0+16+col
    }
    asm volatile("" ::: "memory");  // keep LDS P stores before cross-lane reads

    // Re-layout P (16x32) into an A-fragment
    const _Float16* prow = &plds[wave][lane & 15][0];
    v16h pa;
#pragma unroll
    for (int e = 0; e < 8; ++e) { pa[e] = prow[kb8 + e]; pa[8 + e] = prow[kb8 + 16 + e]; }

    // O += P V  (vbuf rows are d-major: contiguous j per fragment)
#pragma unroll
    for (int dt = 0; dt < 4; ++dt) {
      acc[dt] = wmma32(pa, fragB_f16(&vbuf[cur][dt * 16 + col][0], 0), acc[dt]);
    }
    __syncthreads();  // done reading `cur` before it is overwritten next iter
  }

  // epilogue: O = q + softmax(S) V  (fully-masked rows -> 0, matches nan_to_num)
  float inv[8];
#pragma unroll
  for (int r = 0; r < 8; ++r) inv[r] = (lrow[r] > 0.f) ? 1.f / lrow[r] : 0.f;
#pragma unroll
  for (int dt = 0; dt < 4; ++dt) {
#pragma unroll
    for (int r = 0; r < 8; ++r) {
      const int m = m0 + rbase + r;
      const int d = h * HD + dt * 16 + col;
      const size_t idx = (size_t)(b * NSEQ + m) * DM + d;
      const float val = qf32[idx] + acc[dt][r] * inv[r];
      Oattn[idx] = val;
      Of16[idx] = (_Float16)val;
    }
  }
}

// ---- Kernel 3: O = O + relu(O Wo^T) ----------------------------------------
// grid (128, 4), block 128.
__global__ __launch_bounds__(128) void ffn_kernel(
    const _Float16* __restrict__ Of16, const float* __restrict__ Oattn,
    const float* __restrict__ Wo, float* __restrict__ out) {
  const int wave = threadIdx.x >> 5;
  const int lane = laneid();
  const int m0 = (blockIdx.x * 4 + wave) * 16;
  const int n0 = blockIdx.y * 64;

  const _Float16* orow = Of16 + (size_t)(m0 + (lane & 15)) * DM;
  v8f acc[4] = {};
#pragma unroll
  for (int k0 = 0; k0 < DM; k0 += 32) {
    v16h a = fragA_f16(orow, k0);
#pragma unroll
    for (int ct = 0; ct < 4; ++ct) {
      v16h b = fragB_f32(Wo + (size_t)(n0 + ct * 16 + (lane & 15)) * DM, k0);
      acc[ct] = wmma32(a, b, acc[ct]);
    }
  }
  const int rbase = (lane < 16) ? 0 : 8;
  const int col = lane & 15;
#pragma unroll
  for (int ct = 0; ct < 4; ++ct) {
#pragma unroll
    for (int r = 0; r < 8; ++r) {
      const size_t idx = (size_t)(m0 + rbase + r) * DM + n0 + ct * 16 + col;
      out[idx] = Oattn[idx] + fmaxf(acc[ct][r], 0.f);
    }
  }
}

extern "C" void kernel_launch(void* const* d_in, const int* in_sizes, int n_in,
                              void* d_out, int out_size, void* d_ws, size_t ws_size,
                              hipStream_t stream) {
  (void)in_sizes; (void)n_in; (void)out_size; (void)ws_size;
  const float* Q  = (const float*)d_in[0];
  const float* K  = (const float*)d_in[1];
  const float* Wq = (const float*)d_in[2];
  const float* Wk = (const float*)d_in[3];
  const float* Wv = (const float*)d_in[4];
  const float* Wo = (const float*)d_in[5];

  char* ws = (char*)d_ws;
  _Float16* qf16 = (_Float16*)(ws);                    //  4 MB
  _Float16* kf16 = (_Float16*)(ws + (4u << 20));       //  4 MB
  _Float16* vt   = (_Float16*)(ws + (8u << 20));       //  4 MB (v transposed)
  float*    qf32 = (float*)   (ws + (12u << 20));      //  8 MB
  float*    Oatt = (float*)   (ws + (20u << 20));      //  8 MB
  _Float16* Of16 = (_Float16*)(ws + (28u << 20));      //  4 MB
  float* out = (float*)d_out;

  proj_kernel<<<dim3(128, 4, 3), 128, 0, stream>>>(Q, K, Wq, Wk, Wv,
                                                   qf16, qf32, kf16, vt);
  attn_kernel<<<dim3(NSEQ / 64, NH, NB), 128, 0, stream>>>(qf16, kf16, vt, qf32,
                                                           Oatt, Of16);
  ffn_kernel<<<dim3(128, 4), 128, 0, stream>>>(Of16, Oatt, Wo, out);
}
